// Head_34746285425245
// MI455X (gfx1250) — matrix operand order repaired and checked by reading, not compiled, not run on hardware
//
#include <hip/hip_runtime.h>

typedef float  v8f   __attribute__((ext_vector_type(8)));
typedef float  f32x8 __attribute__((ext_vector_type(8)));
typedef __bf16 v8bf  __attribute__((ext_vector_type(8)));
typedef __bf16 v16bf __attribute__((ext_vector_type(16)));

#define B_  8
#define T_  4096
#define D_  1024
#define DK_ 64
#define L2E 1.4426950408889634f

static __device__ __forceinline__ v16bf cat8(v8bf lo, v8bf hi) {
  return __builtin_shufflevector(lo, hi, 0,1,2,3,4,5,6,7,8,9,10,11,12,13,14,15);
}

static __device__ __forceinline__ v8f wmma_bf16(v16bf a, v16bf b, v8f c) {
  return __builtin_amdgcn_wmma_f32_16x16x32_bf16(false, a, false, b, (short)0, c,
                                                 false, false);
}

// ---------------------------------------------------------------------------
// Kernel 0: convert Wq/Wk/Wv (f32 [1024,64]) to bf16, transposed [64][1024], packed wb[3]
// ---------------------------------------------------------------------------
__global__ __launch_bounds__(256)
void cvt_w_kernel(const float* __restrict__ Wq, const float* __restrict__ Wk,
                  const float* __restrict__ Wv, __bf16* __restrict__ wb) {
  int i = blockIdx.x * 256 + threadIdx.x;           // 0 .. 65535
  if (i < D_ * DK_) {
    wb[(i % DK_) * D_ + i / DK_]               = (__bf16)Wq[i];
    wb[D_ * DK_ + (i % DK_) * D_ + i / DK_]    = (__bf16)Wk[i];
    wb[2 * D_ * DK_ + (i % DK_) * D_ + i / DK_] = (__bf16)Wv[i];
  }
}

// ---------------------------------------------------------------------------
// Kernel 1: QKV projection. One wave per 16 rows of x ([B*T,1024]).
//   qb: bf16 [B*T][64], pre-scaled by 1/8     (row-major -> A-operand loads)
//   kt: bf16 [B*T][64]   (row-major           -> B-operand for q@k^T)
//   vb: bf16 [B][64][T]  (transposed          -> B-operand for P@V)
// ---------------------------------------------------------------------------
__global__ __launch_bounds__(32)
void proj_qkv_kernel(const float* __restrict__ x, const __bf16* __restrict__ wb,
                     __bf16* __restrict__ qb, __bf16* __restrict__ kt,
                     __bf16* __restrict__ vb) {
  const int lane = threadIdx.x;
  const int ln = lane & 15, h = lane >> 4;
  const long t0 = (long)blockIdx.x * 16;
  const float* xrow = x + (t0 + ln) * D_;

  v8f zero = {};
  v8f acc[12];
  #pragma unroll
  for (int i = 0; i < 12; ++i) acc[i] = zero;

  for (int k0 = 0; k0 < D_; k0 += 32) {
    // A tile: lane = row ln, K groups at k0+h*8 and k0+16+h*8 (8 f32 each)
    f32x8 a0 = *(const f32x8*)(xrow + k0 + h * 8);
    f32x8 a1 = *(const f32x8*)(xrow + k0 + 16 + h * 8);
    v16bf A;
    #pragma unroll
    for (int i = 0; i < 8; ++i) { A[i] = (__bf16)a0[i]; A[i + 8] = (__bf16)a1[i]; }

    // B tiles: lane = column ln, K groups at k0+h*8 and k0+16+h*8
    const __bf16* wrow0 = wb + (size_t)ln * D_ + k0 + h * 8;
    #pragma unroll
    for (int w = 0; w < 3; ++w) {
      const __bf16* wrow = wrow0 + (size_t)w * D_ * DK_;
      #pragma unroll
      for (int ct = 0; ct < 4; ++ct) {
        v16bf Bv = cat8(*(const v8bf*)(wrow + (size_t)ct * 16 * D_), *(const v8bf*)(wrow + (size_t)ct * 16 * D_ + 16));
        acc[w * 4 + ct] = wmma_bf16(A, Bv, acc[w * 4 + ct]);
      }
    }
  }

  const long b = t0 >> 12;
  const int tt0 = (int)(t0 & (T_ - 1));
  #pragma unroll
  for (int ct = 0; ct < 4; ++ct) {
    #pragma unroll
    for (int r = 0; r < 8; ++r) {
      const int mrow = r + 8 * h;        // C/D layout: row = r + 8*(lane>>4)
      const int n = ct * 16 + ln;        //             col = lane & 15
      const long trow = t0 + mrow;
      qb[trow * DK_ + n] = (__bf16)(acc[ct][r] * 0.125f);         // q * scale
      kt[trow * DK_ + n] = (__bf16)acc[4 + ct][r];
      vb[(b * DK_ + n) * T_ + tt0 + mrow] = (__bf16)acc[8 + ct][r];
    }
  }
}

// ---------------------------------------------------------------------------
// Kernel 2: causal flash attention. One wave per 16 queries.
// ---------------------------------------------------------------------------
__global__ __launch_bounds__(32)
void attn_fwd_kernel(const __bf16* __restrict__ qb, const __bf16* __restrict__ kt,
                     const __bf16* __restrict__ vb, float* __restrict__ out) {
  const int lane = threadIdx.x;
  const int ln = lane & 15, h = lane >> 4;
  const long t0 = (long)blockIdx.x * 16;
  const long b = t0 >> 12;
  const int tt0 = (int)(t0 & (T_ - 1));

  __shared__ __bf16 pl[16 * 32] __attribute__((aligned(32)));

  // Q tiles in A layout (head-dim 0..31 and 32..63), loaded once
  const __bf16* qrow = qb + (t0 + ln) * DK_;
  v16bf QA0 = cat8(*(const v8bf*)(qrow + h * 8),      *(const v8bf*)(qrow + 16 + h * 8));
  v16bf QA1 = cat8(*(const v8bf*)(qrow + 32 + h * 8), *(const v8bf*)(qrow + 48 + h * 8));

  v8f zero = {};
  v8f acc[4];
  #pragma unroll
  for (int i = 0; i < 4; ++i) acc[i] = zero;
  float mrow[8], lrow[8];
  #pragma unroll
  for (int r = 0; r < 8; ++r) { mrow[r] = -1e30f; lrow[r] = 0.f; }

  const __bf16* ktb = kt + b * DK_ * T_;
  const __bf16* vbb = vb + b * (long)T_ * DK_;
  const int smax = tt0 + 15;

  for (int s0 = 0; s0 <= smax; s0 += 32) {
    // ---- scores S = q*scale @ k^T  (two 16-key N-tiles) -------------------
    const __bf16* k0p = ktb + (size_t)(s0 + ln) * DK_ + h * 8;      // keys s0..s0+15
    const __bf16* k1p = ktb + (size_t)(s0 + 16 + ln) * DK_ + h * 8; // keys s0+16..s0+31
    v8f S0 = wmma_bf16(QA0, cat8(*(const v8bf*)(k0p), *(const v8bf*)(k0p + 16)), zero);
    S0     = wmma_bf16(QA1, cat8(*(const v8bf*)(k0p + 32), *(const v8bf*)(k0p + 48)), S0);
    v8f S1 = wmma_bf16(QA0, cat8(*(const v8bf*)(k1p), *(const v8bf*)(k1p + 16)), zero);
    S1     = wmma_bf16(QA1, cat8(*(const v8bf*)(k1p + 32), *(const v8bf*)(k1p + 48)), S1);

    // ---- causal mask + online softmax ------------------------------------
    #pragma unroll
    for (int r = 0; r < 8; ++r) {
      const int qi = tt0 + r + 8 * h;          // query row of this C element
      float s0v = ((s0 + ln) <= qi)      ? S0[r] : -1e30f;
      float s1v = ((s0 + 16 + ln) <= qi) ? S1[r] : -1e30f;
      float mx = fmaxf(s0v, s1v);
      mx = fmaxf(mx, __shfl_xor(mx, 1, 32));
      mx = fmaxf(mx, __shfl_xor(mx, 2, 32));
      mx = fmaxf(mx, __shfl_xor(mx, 4, 32));
      mx = fmaxf(mx, __shfl_xor(mx, 8, 32));
      float mnew = fmaxf(mrow[r], mx);
      float alpha = __builtin_amdgcn_exp2f((mrow[r] - mnew) * L2E);
      mrow[r] = mnew;
      float p0 = __builtin_amdgcn_exp2f((s0v - mnew) * L2E);
      float p1 = __builtin_amdgcn_exp2f((s1v - mnew) * L2E);
      float rs = p0 + p1;
      rs += __shfl_xor(rs, 1, 32);
      rs += __shfl_xor(rs, 2, 32);
      rs += __shfl_xor(rs, 4, 32);
      rs += __shfl_xor(rs, 8, 32);
      lrow[r] = lrow[r] * alpha + rs;
      #pragma unroll
      for (int ct = 0; ct < 4; ++ct) acc[ct][r] *= alpha;
      // stage P (C/D layout -> LDS, row-major bf16 16x32 tile)
      const int row = r + 8 * h;
      pl[row * 32 + ln] = (__bf16)p0;
      pl[row * 32 + 16 + ln] = (__bf16)p1;
    }
    __syncthreads();
    // re-load P in A layout (lane = query row, K groups h*8 / 16+h*8)
    v16bf PA = cat8(*(const v8bf*)(pl + ln * 32 + h * 8),
                    *(const v8bf*)(pl + ln * 32 + 16 + h * 8));
    __syncthreads();

    // ---- O += P @ V  (B operand: lane = dim ln, key groups h*8 / 16+h*8) -
    const __bf16* vrow = vbb + (size_t)ln * T_ + s0 + h * 8;
    #pragma unroll
    for (int ct = 0; ct < 4; ++ct) {
      v16bf VB = cat8(*(const v8bf*)(vrow + (size_t)ct * 16 * T_), *(const v8bf*)(vrow + (size_t)ct * 16 * T_ + 16));
      acc[ct] = wmma_bf16(PA, VB, acc[ct]);
    }
  }

  // ---- epilogue: O / l ----------------------------------------------------
  #pragma unroll
  for (int ct = 0; ct < 4; ++ct) {
    #pragma unroll
    for (int r = 0; r < 8; ++r) {
      out[(t0 + r + 8 * h) * DK_ + ct * 16 + ln] = acc[ct][r] / lrow[r];
    }
  }
}

// ---------------------------------------------------------------------------
extern "C" void kernel_launch(void* const* d_in, const int* in_sizes, int n_in,
                              void* d_out, int out_size, void* d_ws, size_t ws_size,
                              hipStream_t stream) {
  const float* x  = (const float*)d_in[0];
  const float* Wq = (const float*)d_in[1];
  const float* Wk = (const float*)d_in[2];
  const float* Wv = (const float*)d_in[3];
  float* out = (float*)d_out;

  char* ws = (char*)d_ws;
  const size_t wbBytes = 3ull * D_ * DK_ * sizeof(__bf16);     // 384 KB
  const size_t qkvBytes = (size_t)B_ * T_ * DK_ * sizeof(__bf16); // 4 MB each
  __bf16* wb = (__bf16*)ws;
  __bf16* qb = (__bf16*)(ws + wbBytes);
  __bf16* kt = (__bf16*)(ws + wbBytes + qkvBytes);
  __bf16* vb = (__bf16*)(ws + wbBytes + 2 * qkvBytes);

  hipLaunchKernelGGL(cvt_w_kernel, dim3((D_ * DK_ + 255) / 256), dim3(256), 0,
                     stream, Wq, Wk, Wv, wb);
  hipLaunchKernelGGL(proj_qkv_kernel, dim3(B_ * T_ / 16), dim3(32), 0, stream,
                     x, wb, qb, kt, vb);
  hipLaunchKernelGGL(attn_fwd_kernel, dim3(B_ * T_ / 16), dim3(32), 0, stream,
                     qb, kt, vb, out);
}
